// CIGraphNN_22265110462464
// MI455X (gfx1250) — matrix operand order repaired and checked by reading, not compiled, not run on hardware
//
#include <hip/hip_runtime.h>

// ---------------- WMMA types (CDNA5 wave32) ----------------
typedef __attribute__((ext_vector_type(16))) __bf16 v16bf;
typedef __attribute__((ext_vector_type(8)))  float  v8f;

#define NFEAT 128

// ---------------------------------------------------------------------------
// Weight packing: W is (out=128, in=128) row-major f32.  For D = A(16xK) x
// B(Kx16) with B[k][n] = W[colTile*16+n][k], pack into the wave32 bf16
// B-fragment layout: frag index ((ct*4+ks)*32 + lane)*16 + j, so each lane
// loads one contiguous 32-byte v16bf.
// ---------------------------------------------------------------------------
__global__ void pack_weight_kernel(const float* __restrict__ W, __bf16* __restrict__ pk) {
    int t = blockIdx.x * blockDim.x + threadIdx.x;   // 0..16383
    int j    = t & 15;          // element s: k = 8*(lane>>4) + (s&7) + 16*(s>>3)
    int lane = (t >> 4) & 31;   // lane: n = lane & 15
    int ks   = (t >> 9) & 3;    // k-step
    int ct   = t >> 11;         // column tile 0..7
    pk[t] = (__bf16)W[(ct * 16 + (lane & 15)) * NFEAT + ks * 32 + (lane >> 4) * 8 + (j >> 3) * 16 + (j & 7)];
}

// ---------------------------------------------------------------------------
// Edge scatter-add: agg[dst] += x[src].  One thread per (edge, float4 chunk).
// ---------------------------------------------------------------------------
__global__ void scatter_add_kernel(const float* __restrict__ x,
                                   const int* __restrict__ src,
                                   const int* __restrict__ dst,
                                   float* __restrict__ agg, int E) {
    long long t = (long long)blockIdx.x * blockDim.x + threadIdx.x;
    long long total = (long long)E * 32;
    if (t >= total) return;
    int e  = (int)(t >> 5);
    int c4 = (int)(t & 31);
    int s = src[e], d = dst[e];
    const float4 v = *(const float4*)(x + (size_t)s * NFEAT + c4 * 4);
    float* o = agg + (size_t)d * NFEAT + c4 * 4;
    atomicAdd(o + 0, v.x);
    atomicAdd(o + 1, v.y);
    atomicAdd(o + 2, v.z);
    atomicAdd(o + 3, v.w);
}

// ---------------------------------------------------------------------------
// A-fragment loader: 16-bit A 16x32 layout.
// lanes 0-15: M=lane,   K = kb+[0..7] and kb+16+[0..7]
// lanes16-31: M=lane-16, K = kb+8+[0..7] and kb+24+[0..7]
// rowPtr points at the f32 row; converts f32 -> bf16 on the fly.
// ---------------------------------------------------------------------------
__device__ inline v16bf load_a_frag(const float* __restrict__ rowPtr, int kb, int kHalf) {
    const float4 p0 = *(const float4*)(rowPtr + kb + kHalf);
    const float4 p1 = *(const float4*)(rowPtr + kb + kHalf + 4);
    const float4 p2 = *(const float4*)(rowPtr + kb + kHalf + 16);
    const float4 p3 = *(const float4*)(rowPtr + kb + kHalf + 20);
    v16bf a;
    a[0]  = (__bf16)p0.x; a[1]  = (__bf16)p0.y; a[2]  = (__bf16)p0.z; a[3]  = (__bf16)p0.w;
    a[4]  = (__bf16)p1.x; a[5]  = (__bf16)p1.y; a[6]  = (__bf16)p1.z; a[7]  = (__bf16)p1.w;
    a[8]  = (__bf16)p2.x; a[9]  = (__bf16)p2.y; a[10] = (__bf16)p2.z; a[11] = (__bf16)p2.w;
    a[12] = (__bf16)p3.x; a[13] = (__bf16)p3.y; a[14] = (__bf16)p3.z; a[15] = (__bf16)p3.w;
    return a;
}

__device__ inline v16bf load_b_frag(const __bf16* __restrict__ pk, int ct, int ks, int lane) {
    return *(const v16bf*)(pk + ((((ct * 4 + ks) * 32) + lane) << 4));
}

// ---------------------------------------------------------------------------
// Fused layer kernel.
// MODE 0 (CIConv): Y = relu( L2norm_row( (AGG@Wl^T+bl) * (X@Wr^T+br) ) )
// MODE 1 (Linear): Y = relu( AGG@Wl^T + bl )
// Also accumulates per-column sum / sumsq of Y for the following BatchNorm.
// Block = 256 threads = 8 waves; block owns 16 rows; wave w owns cols
// [16w,16w+16); K=128 -> 4 x v_wmma_f32_16x16x32_bf16 per matrix.
// ---------------------------------------------------------------------------
template <int MODE>
__global__ __launch_bounds__(256)
void ciconv_gemm_kernel(const float* __restrict__ AGG, const float* __restrict__ X,
                        const __bf16* __restrict__ pkL, const __bf16* __restrict__ pkR,
                        const float* __restrict__ bl, const float* __restrict__ br,
                        float* __restrict__ Y,
                        float* __restrict__ bnSum, float* __restrict__ bnSqs, int N) {
    __shared__ float tile[16][NFEAT + 1];
    __shared__ float rnorm[16];

    const int tid   = threadIdx.x;
    const int lane  = tid & 31;
    const int wv    = tid >> 5;          // column tile 0..7
    const int rowBase = blockIdx.x * 16;
    const int laneM = lane & 15;
    const int kHalf = (lane >> 4) * 8;

    v8f accL = {0.f, 0.f, 0.f, 0.f, 0.f, 0.f, 0.f, 0.f};
    v8f accR = {0.f, 0.f, 0.f, 0.f, 0.f, 0.f, 0.f, 0.f};

    const float* aggRow = AGG + (size_t)(rowBase + laneM) * NFEAT;
    const float* xRow   = X   + (size_t)(rowBase + laneM) * NFEAT;

#pragma unroll
    for (int ks = 0; ks < 4; ++ks) {
        const int kb = ks * 32;
        v16bf aL = load_a_frag(aggRow, kb, kHalf);
        v16bf bL = load_b_frag(pkL, wv, ks, lane);
        accL = __builtin_amdgcn_wmma_f32_16x16x32_bf16(false, aL, false, bL,
                                                       (short)0, accL, false, false);
        if (MODE == 0) {
            v16bf aR = load_a_frag(xRow, kb, kHalf);
            v16bf bR = load_b_frag(pkR, wv, ks, lane);
            accR = __builtin_amdgcn_wmma_f32_16x16x32_bf16(false, aR, false, bR,
                                                           (short)0, accR, false, false);
        }
    }

    // C/D layout: VGPR r -> row r (lanes 0-15) / row r+8 (lanes 16-31); col = lane&15.
    const int col   = wv * 16 + laneM;
    const int rOff  = (lane >> 4) * 8;
    const float biasL = bl[col];
    const float biasR = (MODE == 0) ? br[col] : 0.f;
#pragma unroll
    for (int r = 0; r < 8; ++r) {
        float v = accL[r] + biasL;
        if (MODE == 0) v *= (accR[r] + biasR);
        tile[rOff + r][col] = v;
    }
    __syncthreads();

    if (MODE == 0) {
        if (tid < 16) {
            float s = 0.f;
#pragma unroll 8
            for (int c = 0; c < NFEAT; ++c) { float v = tile[tid][c]; s += v * v; }
            rnorm[tid] = rsqrtf(fmaxf(s, 1e-24f));   // 1/max(||row||,1e-12)
        }
        __syncthreads();
    }

    // Write ReLU output + per-column BN partial stats (2 threads per column).
    const int colW  = tid & 127;
    const int rB    = (tid >> 7) * 8;
    float s = 0.f, q = 0.f;
#pragma unroll
    for (int r = 0; r < 8; ++r) {
        int row = rB + r;
        float v = tile[row][colW];
        if (MODE == 0) v *= rnorm[row];
        v = fmaxf(v, 0.f);
        if (rowBase + row < N) Y[(size_t)(rowBase + row) * NFEAT + colW] = v;
        s += v; q += v * v;
    }
    atomicAdd(bnSum + colW, s);
    atomicAdd(bnSqs + colW, q);
}

// ---------------------------------------------------------------------------
// BN finalize: scale/shift from column sums (biased variance, eps=1e-5).
// ---------------------------------------------------------------------------
__global__ void bn_finalize_kernel(const float* __restrict__ sum, const float* __restrict__ sqs,
                                   const float* __restrict__ g, const float* __restrict__ beta,
                                   float* __restrict__ sa, float* __restrict__ sb, float invN) {
    int c = threadIdx.x;
    float m  = sum[c] * invN;
    float v  = fmaxf(sqs[c] * invN - m * m, 0.f);
    float is = rsqrtf(v + 1e-5f);
    float a  = g[c] * is;
    sa[c] = a;
    sb[c] = beta[c] - m * a;
}

__global__ void bn_apply_kernel(const float4* __restrict__ Yin, const float* __restrict__ sa,
                                const float* __restrict__ sb, float4* __restrict__ Xout, int n4) {
    int t = blockIdx.x * blockDim.x + threadIdx.x;
    if (t >= n4) return;
    int c = (t & 31) * 4;                 // 32 float4 per 128-wide row
    float4 y = Yin[t];
    float4 o;
    o.x = fmaf(y.x, sa[c + 0], sb[c + 0]);
    o.y = fmaf(y.y, sa[c + 1], sb[c + 1]);
    o.z = fmaf(y.z, sa[c + 2], sb[c + 2]);
    o.w = fmaf(y.w, sa[c + 3], sb[c + 3]);
    Xout[t] = o;
}

// ---------------------------------------------------------------------------
// Head: logits[i] = relu(x[i,:]·w3 + b3)
// ---------------------------------------------------------------------------
__global__ void logits_kernel(const float* __restrict__ X, const float* __restrict__ w3,
                              const float* __restrict__ b3, float* __restrict__ L, int n) {
    int i = blockIdx.x * blockDim.x + threadIdx.x;
    if (i >= n) return;
    const float4* xp = (const float4*)(X + (size_t)i * NFEAT);
    const float4* wp = (const float4*)w3;
    float s = 0.f;
#pragma unroll
    for (int k = 0; k < 32; ++k) {
        float4 a = xp[k], b = wp[k];
        s = fmaf(a.x, b.x, fmaf(a.y, b.y, fmaf(a.z, b.z, fmaf(a.w, b.w, s))));
    }
    s += b3[0];
    L[i] = fmaxf(s, 0.f);
}

// ---------------------------------------------------------------------------
// Softmax over nodes. red layout: [0..255] block maxima, [256]=gmax, [257]=gsum.
// ---------------------------------------------------------------------------
__global__ void reduce_max_kernel(const float* __restrict__ L, float* __restrict__ red, int n) {
    __shared__ float sm[256];
    float m = -3.0e38f;
    for (int i = blockIdx.x * blockDim.x + threadIdx.x; i < n; i += gridDim.x * blockDim.x)
        m = fmaxf(m, L[i]);
    sm[threadIdx.x] = m; __syncthreads();
    for (int s = 128; s > 0; s >>= 1) {
        if (threadIdx.x < s) sm[threadIdx.x] = fmaxf(sm[threadIdx.x], sm[threadIdx.x + s]);
        __syncthreads();
    }
    if (threadIdx.x == 0) red[blockIdx.x] = sm[0];
}

__global__ void final_max_kernel(float* __restrict__ red) {
    __shared__ float sm[256];
    sm[threadIdx.x] = red[threadIdx.x]; __syncthreads();
    for (int s = 128; s > 0; s >>= 1) {
        if (threadIdx.x < s) sm[threadIdx.x] = fmaxf(sm[threadIdx.x], sm[threadIdx.x + s]);
        __syncthreads();
    }
    if (threadIdx.x == 0) { red[256] = sm[0]; red[257] = 0.f; }
}

__global__ void exp_sum_kernel(float* __restrict__ L, float* __restrict__ red, int n) {
    __shared__ float sm[256];
    const float gm = red[256];
    float s = 0.f;
    for (int i = blockIdx.x * blockDim.x + threadIdx.x; i < n; i += gridDim.x * blockDim.x) {
        float e = expf(L[i] - gm);
        L[i] = e;
        s += e;
    }
    sm[threadIdx.x] = s; __syncthreads();
    for (int st = 128; st > 0; st >>= 1) {
        if (threadIdx.x < st) sm[threadIdx.x] += sm[threadIdx.x + st];
        __syncthreads();
    }
    if (threadIdx.x == 0) atomicAdd(red + 257, sm[0]);
}

__global__ void softmax_out_kernel(const float* __restrict__ L, const float* __restrict__ red,
                                   float* __restrict__ out, int n) {
    int i = blockIdx.x * blockDim.x + threadIdx.x;
    if (i >= n) return;
    out[i] = L[i] * (1.0f / red[257]);
}

// ---------------------------------------------------------------------------
extern "C" void kernel_launch(void* const* d_in, const int* in_sizes, int n_in,
                              void* d_out, int out_size, void* d_ws, size_t ws_size,
                              hipStream_t stream) {
    const float* x0  = (const float*)d_in[0];
    const float* wl1 = (const float*)d_in[2];  const float* bl1 = (const float*)d_in[3];
    const float* wr1 = (const float*)d_in[4];  const float* br1 = (const float*)d_in[5];
    const float* g1  = (const float*)d_in[6];  const float* be1 = (const float*)d_in[7];
    const float* wl2 = (const float*)d_in[8];  const float* bl2 = (const float*)d_in[9];
    const float* wr2 = (const float*)d_in[10]; const float* br2 = (const float*)d_in[11];
    const float* g2  = (const float*)d_in[12]; const float* be2 = (const float*)d_in[13];
    const float* w2  = (const float*)d_in[14]; const float* b2  = (const float*)d_in[15];
    const float* g3  = (const float*)d_in[16]; const float* be3 = (const float*)d_in[17];
    const float* w3  = (const float*)d_in[18]; const float* b3  = (const float*)d_in[19];
    const int*   ei  = (const int*)d_in[20];

    const int N = in_sizes[0] / NFEAT;
    const int E = in_sizes[20] / 2;
    const int* src = ei;
    const int* dst = ei + E;

    // ---- workspace carve-up (256B aligned) ----
    char* ws = (char*)d_ws;
    size_t off = 0;
    auto take = [&](size_t bytes) -> char* {
        char* p = ws + off;
        off = (off + bytes + 255) & ~(size_t)255;
        return p;
    };
    const size_t PKB = 128 * 128 * sizeof(__bf16);
    __bf16* pkWl1 = (__bf16*)take(PKB);
    __bf16* pkWr1 = (__bf16*)take(PKB);
    __bf16* pkWl2 = (__bf16*)take(PKB);
    __bf16* pkWr2 = (__bf16*)take(PKB);
    __bf16* pkW2  = (__bf16*)take(PKB);
    float* stSum = (float*)take(128 * sizeof(float));
    float* stSqs = (float*)take(128 * sizeof(float));
    float* stA   = (float*)take(128 * sizeof(float));
    float* stB   = (float*)take(128 * sizeof(float));
    float* red   = (float*)take(260 * sizeof(float));
    float* logit = (float*)take((size_t)N * sizeof(float));
    const size_t MATB = (size_t)N * NFEAT * sizeof(float);
    float* B1 = (float*)take(MATB);   // agg / x2
    float* B2 = (float*)take(MATB);   // pre-BN activations
    float* B3 = (float*)take(MATB);   // x1 / x3
    (void)n_in; (void)ws_size;

    const int n4 = N * (NFEAT / 4);
    const dim3 blk(256);
    const dim3 gPack(64), gScat((unsigned)(((long long)E * 32 + 255) / 256));
    const dim3 gGemm((N + 15) / 16), gApply((n4 + 255) / 256), gNode((N + 255) / 256);
    const float invN = 1.0f / (float)N;

    // ---- pack all weights to bf16 WMMA B-fragment layout ----
    pack_weight_kernel<<<gPack, blk, 0, stream>>>(wl1, pkWl1);
    pack_weight_kernel<<<gPack, blk, 0, stream>>>(wr1, pkWr1);
    pack_weight_kernel<<<gPack, blk, 0, stream>>>(wl2, pkWl2);
    pack_weight_kernel<<<gPack, blk, 0, stream>>>(wr2, pkWr2);
    pack_weight_kernel<<<gPack, blk, 0, stream>>>(w2,  pkW2);

    // ---- layer 1 ----
    hipMemsetAsync(B1, 0, MATB, stream);
    scatter_add_kernel<<<gScat, blk, 0, stream>>>(x0, src, dst, B1, E);
    hipMemsetAsync(stSum, 0, 2 * 128 * sizeof(float), stream); // stSum+stSqs contiguous? not guaranteed
    hipMemsetAsync(stSqs, 0, 128 * sizeof(float), stream);
    ciconv_gemm_kernel<0><<<gGemm, blk, 0, stream>>>(B1, x0, pkWl1, pkWr1, bl1, br1,
                                                     B2, stSum, stSqs, N);
    bn_finalize_kernel<<<1, 128, 0, stream>>>(stSum, stSqs, g1, be1, stA, stB, invN);
    bn_apply_kernel<<<gApply, blk, 0, stream>>>((const float4*)B2, stA, stB, (float4*)B3, n4);

    // ---- layer 2 ----
    hipMemsetAsync(B1, 0, MATB, stream);
    scatter_add_kernel<<<gScat, blk, 0, stream>>>(B3, src, dst, B1, E);
    hipMemsetAsync(stSum, 0, 128 * sizeof(float), stream);
    hipMemsetAsync(stSqs, 0, 128 * sizeof(float), stream);
    ciconv_gemm_kernel<0><<<gGemm, blk, 0, stream>>>(B1, B3, pkWl2, pkWr2, bl2, br2,
                                                     B2, stSum, stSqs, N);
    bn_finalize_kernel<<<1, 128, 0, stream>>>(stSum, stSqs, g2, be2, stA, stB, invN);
    bn_apply_kernel<<<gApply, blk, 0, stream>>>((const float4*)B2, stA, stB, (float4*)B1, n4);

    // ---- MLP head: relu(x2 @ w2^T + b2) -> BN ----
    hipMemsetAsync(stSum, 0, 128 * sizeof(float), stream);
    hipMemsetAsync(stSqs, 0, 128 * sizeof(float), stream);
    ciconv_gemm_kernel<1><<<gGemm, blk, 0, stream>>>(B1, B1, pkW2, pkW2, b2, b2,
                                                     B2, stSum, stSqs, N);
    bn_finalize_kernel<<<1, 128, 0, stream>>>(stSum, stSqs, g3, be3, stA, stB, invN);
    bn_apply_kernel<<<gApply, blk, 0, stream>>>((const float4*)B2, stA, stB, (float4*)B3, n4);

    // ---- final projection + node softmax ----
    logits_kernel<<<gNode, blk, 0, stream>>>(B3, w3, b3, logit, N);
    reduce_max_kernel<<<256, blk, 0, stream>>>(logit, red, N);
    final_max_kernel<<<1, 256, 0, stream>>>(red);
    exp_sum_kernel<<<256, blk, 0, stream>>>(logit, red, N);
    softmax_out_kernel<<<gNode, blk, 0, stream>>>(logit, red, (float*)d_out, N);
}
